// VRNN_77455440216402
// MI455X (gfx1250) — matrix operand order repaired and checked by reading, not compiled, not run on hardware
//
#include <hip/hip_runtime.h>
#include <hip/hip_bf16.h>
#include <math.h>

typedef __attribute__((ext_vector_type(16))) _Float16 v16h;
typedef __attribute__((ext_vector_type(8)))  _Float16 v8h;
typedef __attribute__((ext_vector_type(8)))  float    v8f;

namespace vrnn {
constexpr int T = 28, B = 16384, X = 28, H = 100, Z = 16;
constexpr int HPH = 128;  // padded H in halves (K padded to 128)
constexpr int XPH = 32;   // padded X
constexpr int ZPH = 32;   // padded Z
constexpr float EPSV = 1e-10f;
constexpr int NWAVES = B / 16;  // 1024 tiles of 16 batch rows

// ---- padded f16 weight blocks inside d_ws (offsets in halves) ----
constexpr int OW_PX  = 0;                     // Wpx   [112 x 32]
constexpr int OW_PZ  = OW_PX  + 112 * 32;     // Wpz   [112 x 32]
constexpr int OW_P1  = OW_PZ  + 112 * 32;     // Wp1   [112 x 128]
constexpr int OW_PMU = OW_P1  + 112 * 128;    // Wp_mu [16 x 128]
constexpr int OW_PSG = OW_PMU + 16 * 128;     // Wp_sig[16 x 128]
constexpr int OW_E1  = OW_PSG + 16 * 128;     // We1   [112 x 256] (2 segs)
constexpr int OW_EMU = OW_E1  + 112 * 256;    // We_mu [16 x 128]
constexpr int OW_ESG = OW_EMU + 16 * 128;     // We_sig[16 x 128]
constexpr int OW_D1  = OW_ESG + 16 * 128;     // Wd1   [112 x 256] (2 segs)
constexpr int OW_D2  = OW_D1  + 112 * 256;    // Wd2   [112 x 128]
constexpr int OW_D3  = OW_D2  + 112 * 128;    // Wd3   [32 x 128]
constexpr int OW_IHR = OW_D3  + 32 * 128;     // Wih rows 0..99   [112 x 256]
constexpr int OW_IHU = OW_IHR + 112 * 256;    // Wih rows 100..199[112 x 256]
constexpr int OW_IHN = OW_IHU + 112 * 256;    // Wih rows 200..299[112 x 256]
constexpr int OW_HHR = OW_IHN + 112 * 256;    // Whh rows 0..99   [112 x 128]
constexpr int OW_HHU = OW_HHR + 112 * 128;    // Whh rows 100..199[112 x 128]
constexpr int OW_HHN = OW_HHU + 112 * 128;    // Whh rows 200..299[112 x 128]
constexpr int W_TOTAL = OW_HHN + 112 * 128;   // 234,496 halves = 458 KB

// ---- bias block in LDS (f32 slot offsets) ----
constexpr int OB_PX  = 0;    // 100
constexpr int OB_PZ  = 100;  // 100
constexpr int OB_P1  = 200;  // 100
constexpr int OB_E1  = 300;  // 100
constexpr int OB_D1  = 400;  // 100
constexpr int OB_D2  = 500;  // 100
constexpr int OB_D3  = 600;  // 28 (pad 32)
constexpr int OB_PMU = 632;  // 16
constexpr int OB_PSG = 648;  // 16
constexpr int OB_EMU = 664;  // 16
constexpr int OB_ESG = 680;  // 16
constexpr int OB_R   = 696;  // 100 (bih+bhh, r gate)
constexpr int OB_U   = 796;  // 100 (bih+bhh, u gate)
constexpr int OB_IN  = 896;  // 100 (bih, n gate)
constexpr int OB_HN  = 996;  // 100 (bhh, n gate)
constexpr int NBIAS  = 1120;

// LDS per wave: f32 x-tile + 8 f16 buffers + f32 bias block
constexpr int LDS_XF  = 16 * 32;                             // f32 words
constexpr int LDS_H16 = 6 * 16 * HPH + 16 * XPH + 16 * ZPH;  // halves
constexpr int LDS_BYTES = LDS_XF * 4 + LDS_H16 * 2 + NBIAS * 4;  // 33,152 B
}

// ---------------- fast hardware transcendentals ------------------------------

__device__ __forceinline__ float sigmoidf_(float v) {
  return 1.0f / (1.0f + __expf(-v));
}
__device__ __forceinline__ float softplusf_(float v) {
  return (v > 20.0f) ? v : __logf(1.0f + __expf(v));
}
__device__ __forceinline__ float tanhf_(float v) {
  const float c = fminf(fmaxf(v, -15.0f), 15.0f);
  const float e = __expf(2.0f * c);
  return (e - 1.0f) / (e + 1.0f);
}

// ---------------- fragment builders (wave32, v_wmma_f32_16x16x32_f16) -------

// A: 16x32 f16. Lanes 0-15 M=lane, halves 0..7 -> K=kbase+8g..+7,
// halves 8..15 -> K=kbase+16+8g..+7. Two aligned ds_load_b128.
__device__ __forceinline__ v16h frag_Ah(const _Float16* a, int lda, int kbase) {
  const int lane = (int)(threadIdx.x & 31);
  const int m = lane & 15, g = lane >> 4;
  const _Float16* p = a + m * lda + kbase + 8 * g;
  v8h lo = *(const v8h*)p;
  v8h hi = *(const v8h*)(p + 16);
  return __builtin_shufflevector(lo, hi, 0, 1, 2, 3, 4, 5, 6, 7, 8, 9, 10, 11,
                                 12, 13, 14, 15);
}

// B: 32x16 f16 = W^T tile. Lane L: column n baked into rowptr,
// K = kbase + 8*(L>>4) + i (i<8), kbase + 16 + 8*(L>>4) + i-8 (i>=8): two global_load_b128.
__device__ __forceinline__ v16h frag_Bh(const _Float16* __restrict__ rowptr,
                                        int kbase) {
  const int lane = (int)(threadIdx.x & 31);
  const _Float16* p = rowptr + kbase + ((lane >> 4) << 3);
  v8h lo = *(const v8h*)p;
  v8h hi = *(const v8h*)(p + 16);
  return __builtin_shufflevector(lo, hi, 0, 1, 2, 3, 4, 5, 6, 7, 8, 9, 10, 11,
                                 12, 13, 14, 15);
}

template <int KS>
__device__ __forceinline__ void gemm_acc(v8f& c, const _Float16* act, int lda,
                                         const _Float16* __restrict__ W,
                                         int ldwp, int n) {
  const _Float16* rowptr = W + (size_t)n * ldwp;
#pragma unroll
  for (int s = 0; s < KS; ++s) {
    v16h a = frag_Ah(act, lda, s * 32);
    v16h b = frag_Bh(rowptr, s * 32);
    c = __builtin_amdgcn_wmma_f32_16x16x32_f16(
        /*neg_a=*/false, a, /*neg_b=*/false, b,
        /*c_mod=*/(short)0, c, /*reuse_a=*/false, /*reuse_b=*/false);
  }
}

// dst[16 x N] f16 (stride HPH) = relu(a1 @ Wseg0^T + a2 @ Wseg1^T + b)
template <int N, int KS1, int LDA1, int KS2, int LDA2>
__device__ __forceinline__ void dense_relu_h(_Float16* dst, const _Float16* a1,
                                             const _Float16* a2,
                                             const _Float16* __restrict__ W,
                                             const _Float16* __restrict__ W2,
                                             int ldwp, const float* bias) {
  using namespace vrnn;
  const int lane = (int)(threadIdx.x & 31);
  const int col = lane & 15, g = lane >> 4;
  constexpr int NT = (N + 15) / 16;
#pragma unroll 1
  for (int nt = 0; nt < NT; ++nt) {
    const int n = nt * 16 + col;
    v8f c = {};
    gemm_acc<KS1>(c, a1, LDA1, W, ldwp, n);
    if (KS2 > 0) gemm_acc<(KS2 > 0 ? KS2 : 1)>(c, a2, LDA2, W2, ldwp, n);
    if (n < N) {
      const float bv = bias[n];  // LDS
#pragma unroll
      for (int j = 0; j < 8; ++j) {
        dst[(j + g * 8) * HPH + n] = (_Float16)fmaxf(c[j] + bv, 0.0f);
      }
    }
  }
}

// ---------------- weight pad/convert (f32 -> padded f16 in d_ws) -------------

__global__ void __launch_bounds__(256)
vrnn_pad_convert(const float* __restrict__ src, int srcld, int r0, int c0,
                 int nrows, int ncols, _Float16* __restrict__ dst, int dstld,
                 int dstc0, int segw, int nrowsPad) {
  const int idx = (int)(blockIdx.x * blockDim.x + threadIdx.x);
  const int total = nrowsPad * segw;
  if (idx >= total) return;
  const int r = idx / segw, c = idx % segw;
  const float v =
      (r < nrows && c < ncols) ? src[(size_t)(r0 + r) * srcld + c0 + c] : 0.0f;
  dst[(size_t)r * dstld + dstc0 + c] = (_Float16)v;
}

// ---------------- main persistent-wave kernel --------------------------------

__global__ void __launch_bounds__(32)
vrnn_fused_kernel(const float* __restrict__ x, const float* __restrict__ eps,
                  const float* __restrict__ bpx, const float* __restrict__ bpz,
                  const float* __restrict__ bp1,
                  const float* __restrict__ bp_mu,
                  const float* __restrict__ bp_sig,
                  const float* __restrict__ be1,
                  const float* __restrict__ be_mu,
                  const float* __restrict__ be_sig,
                  const float* __restrict__ bd1, const float* __restrict__ bd2,
                  const float* __restrict__ bd3, const float* __restrict__ bih,
                  const float* __restrict__ bhh,
                  const _Float16* __restrict__ wsw,
                  float* __restrict__ partial) {
  using namespace vrnn;
  extern __shared__ char smem[];
  const int lane = (int)(threadIdx.x & 31);
  const int col = lane & 15, g = lane >> 4, g8 = g * 8;

  float* xbf = (float*)smem;  // 16 x 32 f32 (exact x for NLL)
  _Float16* hA = (_Float16*)(xbf + LDS_XF);
  _Float16* hB = hA + 16 * HPH;   // h ping/pong
  _Float16* pxb = hB + 16 * HPH;  // phi_x
  _Float16* pzb = pxb + 16 * HPH; // phi_z
  _Float16* t0 = pzb + 16 * HPH;  // he / hd1
  _Float16* t1 = t0 + 16 * HPH;   // hp / hd2
  _Float16* xb = t1 + 16 * HPH;   // 16 x 32 f16 x-tile (GEMM input)
  _Float16* zb = xb + 16 * XPH;   // 16 x 32 f16 z-tile
  float* bb = (float*)(zb + 16 * ZPH);  // bias block (NBIAS f32)

  // Zero everything once: h0 = 0 and all K/N padding stays 0 forever.
  {
    uint32_t* w = (uint32_t*)smem;
    const int nwords = LDS_BYTES / 4;
    for (int i = lane; i < nwords; i += 32) w[i] = 0u;
  }

  // Stage all biases to LDS (GRU r/u biases pre-combined). Same-wave LDS
  // ordering makes these visible without a barrier.
  for (int i = lane; i < H; i += 32) {
    bb[OB_PX + i] = bpx[i];
    bb[OB_PZ + i] = bpz[i];
    bb[OB_P1 + i] = bp1[i];
    bb[OB_E1 + i] = be1[i];
    bb[OB_D1 + i] = bd1[i];
    bb[OB_D2 + i] = bd2[i];
    bb[OB_R + i] = bih[i] + bhh[i];
    bb[OB_U + i] = bih[H + i] + bhh[H + i];
    bb[OB_IN + i] = bih[2 * H + i];
    bb[OB_HN + i] = bhh[2 * H + i];
  }
  if (lane < X) bb[OB_D3 + lane] = bd3[lane];
  if (lane < Z) {
    bb[OB_PMU + lane] = bp_mu[lane];
    bb[OB_PSG + lane] = bp_sig[lane];
    bb[OB_EMU + lane] = be_mu[lane];
    bb[OB_ESG + lane] = be_sig[lane];
  }

  // padded-weight block pointers
  const _Float16* Wpx_h = wsw + OW_PX;
  const _Float16* Wpz_h = wsw + OW_PZ;
  const _Float16* Wp1_h = wsw + OW_P1;
  const _Float16* Wpmu_h = wsw + OW_PMU;
  const _Float16* Wpsg_h = wsw + OW_PSG;
  const _Float16* We1_h = wsw + OW_E1;
  const _Float16* Wemu_h = wsw + OW_EMU;
  const _Float16* Wesg_h = wsw + OW_ESG;
  const _Float16* Wd1_h = wsw + OW_D1;
  const _Float16* Wd2_h = wsw + OW_D2;
  const _Float16* Wd3_h = wsw + OW_D3;
  const _Float16* WihR_h = wsw + OW_IHR;
  const _Float16* WihU_h = wsw + OW_IHU;
  const _Float16* WihN_h = wsw + OW_IHN;
  const _Float16* WhhR_h = wsw + OW_HHR;
  const _Float16* WhhU_h = wsw + OW_HHU;
  const _Float16* WhhN_h = wsw + OW_HHN;

  const int brow0 = (int)blockIdx.x * 16;
  float kl_acc = 0.0f, nll_acc = 0.0f;
  _Float16* h = hA;
  _Float16* hn = hB;

#pragma unroll 1
  for (int t = 0; t < T; ++t) {
    const float* xt = x + ((size_t)t * B + brow0) * X;
    const float* et = eps + ((size_t)t * B + brow0) * Z;

    // stage x_t tile (f32 copy for NLL, f16 copy for GEMM); pads stay zero
#pragma unroll
    for (int r = 0; r < 16; ++r) {
      if (lane < X) {
        const float v = xt[r * X + lane];
        xbf[r * XPH + lane] = v;
        xb[r * XPH + lane] = (_Float16)v;
      }
    }
    // prefetch next timestep's inputs toward L2 (global_prefetch_b8)
    if (t + 1 < T) {
      __builtin_prefetch(x + ((size_t)(t + 1) * B + brow0) * X + lane * 4, 0, 1);
      __builtin_prefetch(eps + ((size_t)(t + 1) * B + brow0) * Z + lane * 4, 0, 1);
    }

    // px = relu(x_t @ Wpx^T + bpx)
    dense_relu_h<H, 1, XPH, 0, 0>(pxb, xb, nullptr, Wpx_h, nullptr, 32,
                                  bb + OB_PX);
    // he = relu([px, h] @ We1^T + be1)
    dense_relu_h<H, 4, HPH, 4, HPH>(t0, pxb, h, We1_h, We1_h + 128, 256,
                                    bb + OB_E1);
    // enc_mu / enc_sig linear parts -> registers (N = Z = 16, one tile)
    v8f cmu = {}, csg = {};
    gemm_acc<4>(cmu, t0, HPH, Wemu_h, 128, col);
    gemm_acc<4>(csg, t0, HPH, Wesg_h, 128, col);
    // hp = relu(h @ Wp1^T + bp1)
    dense_relu_h<H, 4, HPH, 0, 0>(t1, h, nullptr, Wp1_h, nullptr, 128,
                                  bb + OB_P1);
    v8f pmu = {}, psg = {};
    gemm_acc<4>(pmu, t1, HPH, Wpmu_h, 128, col);
    gemm_acc<4>(psg, t1, HPH, Wpsg_h, 128, col);

    // reparameterize + KL in registers; z -> f16 LDS
#pragma unroll
    for (int j = 0; j < 8; ++j) {
      const int row = j + g8;
      const float emu = cmu[j] + bb[OB_EMU + col];
      const float esg = softplusf_(csg[j] + bb[OB_ESG + col]);
      const float prm = pmu[j] + bb[OB_PMU + col];
      const float prs = softplusf_(psg[j] + bb[OB_PSG + col]);
      const float ev = et[row * Z + col];
      zb[row * ZPH + col] = (_Float16)(emu + sqrtf(esg) * ev);
      const float ps = prs + EPSV;
      const float dm = emu - prm;
      kl_acc += 0.5f * (2.0f * __logf(ps) - 2.0f * __logf(esg) +
                        (esg * esg + dm * dm) / (ps * ps) - 1.0f);
    }

    // pz = relu(z @ Wpz^T + bpz)
    dense_relu_h<H, 1, ZPH, 0, 0>(pzb, zb, nullptr, Wpz_h, nullptr, 32,
                                  bb + OB_PZ);
    // hd1 = relu([pz, h] @ Wd1^T + bd1)   (reuse t0)
    dense_relu_h<H, 4, HPH, 4, HPH>(t0, pzb, h, Wd1_h, Wd1_h + 128, 256,
                                    bb + OB_D1);
    // hd2 = relu(hd1 @ Wd2^T + bd2)       (reuse t1)
    dense_relu_h<H, 4, HPH, 0, 0>(t1, t0, nullptr, Wd2_h, nullptr, 128,
                                  bb + OB_D2);

    // probs = sigmoid(hd2 @ Wd3^T + bd3), fused Bernoulli NLL (N = X = 28)
#pragma unroll 1
    for (int nt = 0; nt < 2; ++nt) {
      const int n = nt * 16 + col;
      v8f cp = {};
      gemm_acc<4>(cp, t1, HPH, Wd3_h, 128, n);
      if (n < X) {
        const float bv = bb[OB_D3 + n];
#pragma unroll
        for (int j = 0; j < 8; ++j) {
          const int row = j + g8;
          const float p = sigmoidf_(cp[j] + bv);
          const float xv = xbf[row * XPH + n];
          nll_acc -=
              xv * __logf(p + EPSV) + (1.0f - xv) * __logf(1.0f - p + EPSV);
        }
      }
    }

    // GRUCell(input=[pz, px], h): per-gate padded blocks, gates in registers
#pragma unroll 1
    for (int nt = 0; nt < 7; ++nt) {
      const int n = nt * 16 + col;
      v8f cr = {}, cu = {}, ci = {}, chn = {};
      gemm_acc<4>(cr, pzb, HPH, WihR_h, 256, n);
      gemm_acc<4>(cr, pxb, HPH, WihR_h + 128, 256, n);
      gemm_acc<4>(cr, h, HPH, WhhR_h, 128, n);
      gemm_acc<4>(cu, pzb, HPH, WihU_h, 256, n);
      gemm_acc<4>(cu, pxb, HPH, WihU_h + 128, 256, n);
      gemm_acc<4>(cu, h, HPH, WhhU_h, 128, n);
      gemm_acc<4>(ci, pzb, HPH, WihN_h, 256, n);
      gemm_acc<4>(ci, pxb, HPH, WihN_h + 128, 256, n);
      gemm_acc<4>(chn, h, HPH, WhhN_h, 128, n);
      if (n < H) {
        const float br = bb[OB_R + n];
        const float bu = bb[OB_U + n];
        const float bin = bb[OB_IN + n];
        const float bhn = bb[OB_HN + n];
#pragma unroll
        for (int j = 0; j < 8; ++j) {
          const int row = j + g8;
          const float r = sigmoidf_(cr[j] + br);
          const float u = sigmoidf_(cu[j] + bu);
          const float nnv = tanhf_(ci[j] + bin + r * (chn[j] + bhn));
          const float hv = (float)h[row * HPH + n];
          hn[row * HPH + n] = (_Float16)((1.0f - u) * nnv + u * hv);
        }
      }
    }
    // ping-pong h (pads of both buffers stay zero: only cols < H written)
    _Float16* tmp = h;
    h = hn;
    hn = tmp;
  }

  // wave-level reduction of the two loss partials
#pragma unroll
  for (int off = 16; off > 0; off >>= 1) {
    kl_acc += __shfl_xor(kl_acc, off, 32);
    nll_acc += __shfl_xor(nll_acc, off, 32);
  }
  if (lane == 0) {
    partial[2 * (size_t)blockIdx.x + 0] = kl_acc;
    partial[2 * (size_t)blockIdx.x + 1] = nll_acc;
  }
}

// ---------------- deterministic final reduction ------------------------------

__global__ void __launch_bounds__(256)
vrnn_reduce_kernel(const float* __restrict__ partial, int nw,
                   float* __restrict__ out) {
  __shared__ float s0[256], s1[256];
  float a = 0.0f, b = 0.0f;
  for (int i = (int)threadIdx.x; i < nw; i += 256) {
    a += partial[2 * i + 0];
    b += partial[2 * i + 1];
  }
  s0[threadIdx.x] = a;
  s1[threadIdx.x] = b;
  __syncthreads();
  for (int st = 128; st > 0; st >>= 1) {
    if ((int)threadIdx.x < st) {
      s0[threadIdx.x] += s0[threadIdx.x + st];
      s1[threadIdx.x] += s1[threadIdx.x + st];
    }
    __syncthreads();
  }
  if (threadIdx.x == 0) {
    out[0] = s0[0] / (float)vrnn::B;  // sum over T of mean-over-B KL
    out[1] = s1[0] / (float)vrnn::B;  // sum over T of mean-over-B NLL
  }
}

// ---------------- launch ------------------------------------------------------

extern "C" void kernel_launch(void* const* d_in, const int* in_sizes, int n_in,
                              void* d_out, int out_size, void* d_ws,
                              size_t ws_size, hipStream_t stream) {
  (void)in_sizes; (void)n_in; (void)out_size; (void)ws_size;
  using namespace vrnn;
  const float* x      = (const float*)d_in[0];
  const float* eps    = (const float*)d_in[1];
  const float* Wpx    = (const float*)d_in[2];
  const float* bpx    = (const float*)d_in[3];
  const float* Wpz    = (const float*)d_in[4];
  const float* bpz    = (const float*)d_in[5];
  const float* Wp1    = (const float*)d_in[6];
  const float* bp1    = (const float*)d_in[7];
  const float* Wp_mu  = (const float*)d_in[8];
  const float* bp_mu  = (const float*)d_in[9];
  const float* Wp_sig = (const float*)d_in[10];
  const float* bp_sig = (const float*)d_in[11];
  const float* We1    = (const float*)d_in[12];
  const float* be1    = (const float*)d_in[13];
  const float* We_mu  = (const float*)d_in[14];
  const float* be_mu  = (const float*)d_in[15];
  const float* We_sig = (const float*)d_in[16];
  const float* be_sig = (const float*)d_in[17];
  const float* Wd1    = (const float*)d_in[18];
  const float* bd1    = (const float*)d_in[19];
  const float* Wd2    = (const float*)d_in[20];
  const float* bd2    = (const float*)d_in[21];
  const float* Wd3    = (const float*)d_in[22];
  const float* bd3    = (const float*)d_in[23];
  const float* Wih    = (const float*)d_in[24];
  const float* Whh    = (const float*)d_in[25];
  const float* bih    = (const float*)d_in[26];
  const float* bhh    = (const float*)d_in[27];

  _Float16* wsw = (_Float16*)d_ws;                 // padded f16 weights
  float* partial = (float*)(wsw + W_TOTAL);        // NWAVES x 2 floats after

  auto conv = [&](const float* src, int srcld, int r0, int c0, int nr, int nc,
                  int dstOff, int dstld, int dstc0, int segw, int nrPad) {
    const int total = nrPad * segw;
    vrnn_pad_convert<<<(total + 255) / 256, 256, 0, stream>>>(
        src, srcld, r0, c0, nr, nc, wsw + dstOff, dstld, dstc0, segw, nrPad);
  };

  conv(Wpx,    28,    0,   0, 100,  28, OW_PX,  32,    0,  32, 112);
  conv(Wpz,    16,    0,   0, 100,  16, OW_PZ,  32,    0,  32, 112);
  conv(Wp1,   100,    0,   0, 100, 100, OW_P1,  128,   0, 128, 112);
  conv(Wp_mu, 100,    0,   0,  16, 100, OW_PMU, 128,   0, 128,  16);
  conv(Wp_sig,100,    0,   0,  16, 100, OW_PSG, 128,   0, 128,  16);
  conv(We1,   200,    0,   0, 100, 100, OW_E1,  256,   0, 128, 112);
  conv(We1,   200,    0, 100, 100, 100, OW_E1,  256, 128, 128, 112);
  conv(We_mu, 100,    0,   0,  16, 100, OW_EMU, 128,   0, 128,  16);
  conv(We_sig,100,    0,   0,  16, 100, OW_ESG, 128,   0, 128,  16);
  conv(Wd1,   200,    0,   0, 100, 100, OW_D1,  256,   0, 128, 112);
  conv(Wd1,   200,    0, 100, 100, 100, OW_D1,  256, 128, 128, 112);
  conv(Wd2,   100,    0,   0, 100, 100, OW_D2,  128,   0, 128, 112);
  conv(Wd3,   100,    0,   0,  28, 100, OW_D3,  128,   0, 128,  32);
  conv(Wih,   200,    0,   0, 100, 100, OW_IHR, 256,   0, 128, 112);
  conv(Wih,   200,    0, 100, 100, 100, OW_IHR, 256, 128, 128, 112);
  conv(Wih,   200,  100,   0, 100, 100, OW_IHU, 256,   0, 128, 112);
  conv(Wih,   200,  100, 100, 100, 100, OW_IHU, 256, 128, 128, 112);
  conv(Wih,   200,  200,   0, 100, 100, OW_IHN, 256,   0, 128, 112);
  conv(Wih,   200,  200, 100, 100, 100, OW_IHN, 256, 128, 128, 112);
  conv(Whh,   100,    0,   0, 100, 100, OW_HHR, 128,   0, 128, 112);
  conv(Whh,   100,  100,   0, 100, 100, OW_HHU, 128,   0, 128, 112);
  conv(Whh,   100,  200,   0, 100, 100, OW_HHN, 128,   0, 128, 112);

  vrnn_fused_kernel<<<NWAVES, 32, LDS_BYTES, stream>>>(
      x, eps, bpx, bpz, bp1, bp_mu, bp_sig, be1, be_mu, be_sig, bd1, bd2, bd3,
      bih, bhh, wsw, partial);

  vrnn_reduce_kernel<<<1, 256, 0, stream>>>(partial, NWAVES, (float*)d_out);
}
